// HGCN_85186381349133
// MI455X (gfx1250) — matrix operand order repaired and checked by reading, not compiled, not run on hardware
//
#include <hip/hip_runtime.h>
#include <hip/hip_bf16.h>
#include <math.h>

typedef __attribute__((ext_vector_type(16))) _Float16 v16h;
typedef __attribute__((ext_vector_type(8)))  float    v8f;

#define DIM 128
#define HID 64
#define K3D 384
#define EPSV 1e-6f

// Async global->LDS copy of one 16B chunk. lds = workgroup-relative LDS byte
// address (low 32 bits of a generic LDS pointer), g = 64-bit global address.
__device__ __forceinline__ void async_b128(unsigned lds, unsigned long long g) {
    asm volatile("global_load_async_to_lds_b128 %0, %1, off"
                 :: "v"(lds), "v"(g) : "memory");
}
__device__ __forceinline__ void wait_async0() {
    asm volatile("s_wait_asynccnt 0x0" ::: "memory");
}

__device__ __forceinline__ float clamp_curv(float c) {
    return fminf(fmaxf(c, 0.1f), 10.0f);
}

__device__ __forceinline__ void atomicMaxF32(float* addr, float val) {
    int* ai = (int*)addr;
    int old = __float_as_int(*addr);
    while (__int_as_float(old) < val) {
        int assumed = old;
        old = atomicCAS(ai, assumed, __float_as_int(val));
        if (old == assumed) break;
    }
}

// 128-thread block reduction (sum). Safe for repeated use of same scratch.
__device__ __forceinline__ float blockReduceSum128(float v, float* red) {
    int t = threadIdx.x;
    __syncthreads();
    red[t] = v;
    __syncthreads();
#pragma unroll
    for (int s = 64; s > 0; s >>= 1) {
        if (t < s) red[t] += red[t + s];
        __syncthreads();
    }
    return red[0];
}

// ---------------------------------------------------------------- f32 -> f16
__global__ void cvt_f16_kernel(const float* __restrict__ s, _Float16* __restrict__ d, int n) {
    int i = blockIdx.x * 256 + threadIdx.x;
    if (i < n) d[i] = (_Float16)s[i];
}

// ------------------------------------------------- init segment scratch/agg
__global__ void init_seg_kernel(float* segmax, float* segsum, float* xagg, int N) {
    int i = blockIdx.x * 256 + threadIdx.x;
    if (i < N) { segmax[i] = -3.0e38f; segsum[i] = 0.0f; }
    if (i < N * DIM) xagg[i] = 0.0f;
}

// ---------------------------------------------------------------- log map
// block = 128 threads, one node per block
__global__ void logmap_kernel(const float* __restrict__ xh, float* __restrict__ xt,
                              _Float16* __restrict__ xtf, const float* __restrict__ curv_p) {
    __shared__ float red[128];
    int node = blockIdx.x;
    int t = threadIdx.x;
    float c  = clamp_curv(curv_p[0]);
    float sc = sqrtf(c);
    const float* row = xh + (size_t)node * (DIM + 1);
    float x0   = fmaxf(sc * row[0], 1.0f + 1e-7f);
    float dist = acoshf(x0) / sc;
    float v = row[1 + t];
    float ss = blockReduceSum128(v * v, red);
    float nrm = fmaxf(sqrtf(ss), EPSV);
    float o = v * (dist / nrm);
    xt [(size_t)node * DIM + t] = o;
    xtf[(size_t)node * DIM + t] = (_Float16)o;
}

// ------------------------------------------------- node GEMM: x_lin = x_tan @ lin_w^T + b
// block = 256 threads (8 waves), one 16-row tile per block; wave w -> col tile w
__global__ void node_gemm_kernel(const _Float16* __restrict__ A, const _Float16* __restrict__ W,
                                 const float* __restrict__ bias, float* __restrict__ xl,
                                 _Float16* __restrict__ xlf, int N) {
    __shared__ __align__(16) _Float16 at[16 * DIM];  // 4 KB
    int m0 = blockIdx.x * 16;

    // async stage A tile: 256 chunks x 16B, one per thread
    {
        unsigned lbase = (unsigned)(uintptr_t)(void*)&at[0];
        int c   = threadIdx.x;           // 0..255
        int r   = c >> 4;
        int off = (c & 15) * 16;         // byte offset within 256B row
        int gm  = m0 + r;
        if (gm < N) {
            unsigned long long g =
                (unsigned long long)(uintptr_t)(A + (size_t)gm * DIM) + (unsigned)off;
            async_b128(lbase + (unsigned)(c * 16), g);
        } else {
#pragma unroll
            for (int j = 0; j < 8; ++j) at[c * 8 + j] = (_Float16)0.0f;
        }
        wait_async0();
    }
    __syncthreads();

    int wid  = threadIdx.x >> 5;
    int lane = threadIdx.x & 31;
    int n0   = wid * 16;
    int row  = lane & 15;
    int col  = lane & 15;
    int kbA  = (lane < 16) ? 0 : 16;
    int kbB  = (lane < 16) ? 0 : 16;

    v8f acc = {};
#pragma unroll
    for (int ks = 0; ks < DIM / 32; ++ks) {
        v16h a, b;
#pragma unroll
        for (int h = 0; h < 16; ++h) {
            int ka = ks * 32 + kbA + h;
            a[h] = at[row * DIM + ka];
            int kb = ks * 32 + kbB + h;
            b[h] = W[(size_t)(n0 + col) * DIM + kb];  // B[k][col] = lin_w[col][k]
        }
        acc = __builtin_amdgcn_wmma_f32_16x16x32_f16(false, a, false, b, (short)0, acc, false, false);
    }

    int mb = (lane < 16) ? 0 : 8;
#pragma unroll
    for (int r = 0; r < 8; ++r) {
        int m = m0 + mb + r;
        int n = n0 + col;
        if (m < N) {
            float v = acc[r] + bias[n];
            xl [(size_t)m * DIM + n] = v;
            xlf[(size_t)m * DIM + n] = (_Float16)v;
        }
    }
}

// ------------------------------------------------- edge attention MLP + score + seg max
// block = 128 threads (4 waves), persistent grid-stride over 16-edge tiles
__global__ void edge_score_kernel(const _Float16* __restrict__ xlf, const _Float16* __restrict__ embf,
                                  const _Float16* __restrict__ w1f, const float* __restrict__ b1,
                                  const float* __restrict__ w2, const float* __restrict__ b2p,
                                  const float* __restrict__ sibp, const int* __restrict__ srcp,
                                  const int* __restrict__ dstp, const int* __restrict__ et,
                                  const float* __restrict__ ew, float* __restrict__ score,
                                  float* __restrict__ segmax, int E) {
    __shared__ __align__(16) _Float16 w1s[K3D * HID];  // 48 KB
    __shared__ __align__(16) _Float16 at[16 * K3D];    // 12 KB, aliased as float hbuf after WMMA
    __shared__ int   meta[48];                         // dst,src,type for 16 edges
    __shared__ float w2s[HID];
    __shared__ float pred[128];                        // score partial sums

    // async stage w1 once: 48KB contiguous = 3072 x 16B chunks
    {
        unsigned lbase = (unsigned)(uintptr_t)(void*)&w1s[0];
        unsigned long long gbase = (unsigned long long)(uintptr_t)w1f;
        for (int c = threadIdx.x; c < (K3D * HID * 2) / 16; c += 128)
            async_b128(lbase + (unsigned)(c * 16), gbase + (unsigned long long)c * 16);
    }
    if (threadIdx.x < HID) w2s[threadIdx.x] = w2[threadIdx.x];
    float b2  = b2p[0];
    float sib = sibp[0];
    wait_async0();

    unsigned atbase = (unsigned)(uintptr_t)(void*)&at[0];
    int ntiles = (E + 15) / 16;
    for (int tile = blockIdx.x; tile < ntiles; tile += gridDim.x) {
        int e0 = tile * 16;
        __syncthreads();  // protect at/meta reuse (also covers w1s/w2s staging on first iter)
        if (threadIdx.x < 16) {
            int e = e0 + threadIdx.x;
            int ec = (e < E) ? e : (E - 1);
            meta[threadIdx.x]      = dstp[ec];
            meta[16 + threadIdx.x] = srcp[ec];
            meta[32 + threadIdx.x] = et[ec];
        }
        __syncthreads();

        // async-gather attn_in tile: row r = [x_lin[dst] | x_lin[src] | emb[type]],
        // 16 rows x 768B = 768 x 16B chunks, 6 per thread
#pragma unroll
        for (int i = 0; i < 6; ++i) {
            int c   = threadIdx.x + i * 128;   // 0..767
            int r   = c / 48;
            int off = (c % 48) * 16;           // byte offset within 768B row
            int d_  = meta[r], s_ = meta[16 + r], ty = meta[32 + r];
            unsigned long long g;
            if (off < 256)
                g = (unsigned long long)(uintptr_t)(xlf + (size_t)d_ * DIM) + (unsigned)off;
            else if (off < 512)
                g = (unsigned long long)(uintptr_t)(xlf + (size_t)s_ * DIM) + (unsigned)(off - 256);
            else
                g = (unsigned long long)(uintptr_t)(embf + (size_t)ty * DIM) + (unsigned)(off - 512);
            async_b128(atbase + (unsigned)(c * 16), g);
        }
        wait_async0();
        __syncthreads();

        int wid  = threadIdx.x >> 5;
        int lane = threadIdx.x & 31;
        int col  = lane & 15;
        int rw   = lane & 15;
        int kbA  = (lane < 16) ? 0 : 16;
        int kbB  = (lane < 16) ? 0 : 16;

        v8f acc = {};
#pragma unroll
        for (int ks = 0; ks < K3D / 32; ++ks) {
            v16h a, b;
#pragma unroll
            for (int h = 0; h < 16; ++h) {
                int ka = ks * 32 + kbA + h;
                a[h] = at[rw * K3D + ka];
                int kb = ks * 32 + kbB + h;
                b[h] = w1s[kb * HID + wid * 16 + col];
            }
            acc = __builtin_amdgcn_wmma_f32_16x16x32_f16(false, a, false, b, (short)0, acc, false, false);
        }
        __syncthreads();  // done reading `at` -> safe to alias as hbuf

        float* hbuf = (float*)at;  // 16 x 64 f32 (4 KB of the 12 KB region)
        int mb = (lane < 16) ? 0 : 8;
        int cg = wid * 16 + col;
        float b1v = b1[cg];
        float w2v = w2s[cg];
#pragma unroll
        for (int r = 0; r < 8; ++r) {
            float x  = acc[r] + b1v;
            float sg = 1.0f / (1.0f + expf(-x));   // silu
            hbuf[(mb + r) * HID + cg] = x * sg * w2v;
        }
        __syncthreads();

        // parallel reduce over the 64 hidden units: 8 partials per row
        {
            int r  = threadIdx.x >> 3;
            int g2 = threadIdx.x & 7;
            float ps = 0.0f;
#pragma unroll
            for (int j = 0; j < 8; ++j) ps += hbuf[r * HID + g2 * 8 + j];
            pred[threadIdx.x] = ps;
        }
        __syncthreads();

        if (threadIdx.x < 16) {
            int e = e0 + threadIdx.x;
            if (e < E) {
                float sum = b2;
#pragma unroll
                for (int j = 0; j < 8; ++j) sum += pred[threadIdx.x * 8 + j];
                sum += logf(fmaxf(ew[e], EPSV));
                if (meta[32 + threadIdx.x] == 1) sum += sib;   // SIB_ID == 1
                score[e] = sum;
                atomicMaxF32(&segmax[meta[threadIdx.x]], sum);
            }
        }
    }
}

// ------------------------------------------------- exp(score - max), segment sum
__global__ void exp_sum_kernel(const float* __restrict__ score, const int* __restrict__ dstp,
                               const float* __restrict__ segmax, float* __restrict__ exv,
                               float* __restrict__ segsum, int E) {
    int i = blockIdx.x * 256 + threadIdx.x;
    if (i >= E) return;
    int d = dstp[i];
    float v = expf(score[i] - segmax[d]);
    exv[i] = v;
    atomicAdd(&segsum[d], v);
}

// ------------------------------------------------- weighted message scatter-add
// block = 256 threads -> 2 edges x 128 features
__global__ void aggregate_kernel(const float* __restrict__ xl, const float* __restrict__ emb,
                                 const int* __restrict__ srcp, const int* __restrict__ dstp,
                                 const int* __restrict__ et, const float* __restrict__ ew,
                                 const float* __restrict__ exv, const float* __restrict__ segsum,
                                 float* __restrict__ xagg, int E) {
    int e = blockIdx.x * 2 + (threadIdx.x >> 7);
    int f = threadIdx.x & 127;
    if (e >= E) return;
    int d = dstp[e], s = srcp[e], ty = et[e];
    float alpha = exv[e] / (segsum[d] + 1e-16f);
    float coef  = alpha * ew[e];
    float m = (xl[(size_t)s * DIM + f] + emb[ty * DIM + f]) * coef;
    atomicAdd(&xagg[(size_t)d * DIM + f], m);
}

// ------------------------------------------------- residual + LN + exp map
__global__ void finalize_kernel(const float* __restrict__ xt, const float* __restrict__ xagg,
                                const float* __restrict__ g, const float* __restrict__ b,
                                const float* __restrict__ curv_p, float* __restrict__ out) {
    __shared__ float red[128];
    int node = blockIdx.x;
    int t = threadIdx.x;
    float c  = clamp_curv(curv_p[0]);
    float sc = sqrtf(c);

    float x = xt[(size_t)node * DIM + t] + xagg[(size_t)node * DIM + t];
    float mu  = blockReduceSum128(x, red) * (1.0f / DIM);
    float dv  = x - mu;
    float var = blockReduceSum128(dv * dv, red) * (1.0f / DIM);
    float xn  = dv * rsqrtf(var + 1e-5f) * g[t] + b[t];

    float ss  = blockReduceSum128(xn * xn, red);
    float nrm = fmaxf(sqrtf(ss), EPSV);
    float th  = sc * nrm;
    float x0  = coshf(th) / sc;
    float sp  = xn * (sinhf(th) / (sc * nrm));
    out[(size_t)node * (DIM + 1) + 1 + t] = sp;
    if (t == 0) out[(size_t)node * (DIM + 1)] = x0;
}

// ============================================================== host driver
extern "C" void kernel_launch(void* const* d_in, const int* in_sizes, int n_in,
                              void* d_out, int out_size, void* d_ws, size_t ws_size,
                              hipStream_t stream) {
    (void)n_in; (void)out_size; (void)ws_size;
    const float* x_hyp        = (const float*)d_in[0];
    const int*   edge_index   = (const int*)  d_in[1];
    const int*   edge_types   = (const int*)  d_in[2];
    const float* edge_weights = (const float*)d_in[3];
    const float* lin_w        = (const float*)d_in[4];
    const float* lin_b        = (const float*)d_in[5];
    const float* ln_g         = (const float*)d_in[6];
    const float* ln_b         = (const float*)d_in[7];
    const float* edge_emb     = (const float*)d_in[8];
    const float* attn_w1      = (const float*)d_in[9];
    const float* attn_b1      = (const float*)d_in[10];
    const float* attn_w2      = (const float*)d_in[11];
    const float* attn_b2      = (const float*)d_in[12];
    const float* sib          = (const float*)d_in[13];
    const float* curv         = (const float*)d_in[14];

    const int N = in_sizes[0] / (DIM + 1);
    const int E = in_sizes[2];
    const int L = in_sizes[14];
    const int T = in_sizes[8] / (L * DIM);

    char* p = (char*)d_ws;
    auto carve = [&](size_t bytes) -> void* {
        void* r = (void*)p;
        p += (bytes + 255) & ~(size_t)255;
        return r;
    };
    float*    xt     = (float*)   carve((size_t)N * DIM * 4);
    float*    xl     = (float*)   carve((size_t)N * DIM * 4);
    float*    xagg   = (float*)   carve((size_t)N * DIM * 4);
    float*    mid0   = (float*)   carve((size_t)N * (DIM + 1) * 4);
    float*    mid1   = (float*)   carve((size_t)N * (DIM + 1) * 4);
    _Float16* xtf    = (_Float16*)carve((size_t)N * DIM * 2);
    _Float16* xlf    = (_Float16*)carve((size_t)N * DIM * 2);
    _Float16* wlinf  = (_Float16*)carve((size_t)DIM * DIM * 2);
    _Float16* w1f    = (_Float16*)carve((size_t)K3D * HID * 2);
    _Float16* embf   = (_Float16*)carve((size_t)T * DIM * 2);
    float*    score  = (float*)   carve((size_t)E * 4);
    float*    exv    = (float*)   carve((size_t)E * 4);
    float*    segmax = (float*)   carve((size_t)N * 4);
    float*    segsum = (float*)   carve((size_t)N * 4);

    const int* srcp = edge_index;        // edge_index[0] = src
    const int* dstp = edge_index + E;    // edge_index[1] = dst

    const float* cur = x_hyp;
    for (int l = 0; l < L; ++l) {
        cvt_f16_kernel<<<(DIM * DIM + 255) / 256, 256, 0, stream>>>(
            lin_w + (size_t)l * DIM * DIM, wlinf, DIM * DIM);
        cvt_f16_kernel<<<(K3D * HID + 255) / 256, 256, 0, stream>>>(
            attn_w1 + (size_t)l * K3D * HID, w1f, K3D * HID);
        cvt_f16_kernel<<<(T * DIM + 255) / 256, 256, 0, stream>>>(
            edge_emb + (size_t)l * T * DIM, embf, T * DIM);

        init_seg_kernel<<<((size_t)N * DIM + 255) / 256, 256, 0, stream>>>(segmax, segsum, xagg, N);

        logmap_kernel<<<N, 128, 0, stream>>>(cur, xt, xtf, curv + l);

        node_gemm_kernel<<<(N + 15) / 16, 256, 0, stream>>>(
            xtf, wlinf, lin_b + (size_t)l * DIM, xl, xlf, N);

        edge_score_kernel<<<4096, 128, 0, stream>>>(
            xlf, embf, w1f, attn_b1 + (size_t)l * HID, attn_w2 + (size_t)l * HID,
            attn_b2 + l, sib + l, srcp, dstp, edge_types, edge_weights,
            score, segmax, E);

        exp_sum_kernel<<<(E + 255) / 256, 256, 0, stream>>>(score, dstp, segmax, exv, segsum, E);

        aggregate_kernel<<<(E + 1) / 2, 256, 0, stream>>>(
            xl, edge_emb + (size_t)l * T * DIM, srcp, dstp, edge_types, edge_weights,
            exv, segsum, xagg, E);

        float* outp = (l == L - 1) ? (float*)d_out : ((l & 1) ? mid1 : mid0);
        finalize_kernel<<<N, 128, 0, stream>>>(
            xt, xagg, ln_g + (size_t)l * DIM, ln_b + (size_t)l * DIM, curv + l, outp);
        cur = outp;
    }
}
